// Stacked_attention_30176440222184
// MI455X (gfx1250) — matrix operand order repaired and checked by reading, not compiled, not run on hardware
//
#include <hip/hip_runtime.h>

typedef __attribute__((ext_vector_type(16))) _Float16 v16h;
typedef __attribute__((ext_vector_type(8)))  float    v8f;

union FragH { v16h h; float4 f4[2]; };

#define TPB 256

// wait for all outstanding async global->LDS transfers of this wave
#define ASYNC_WAIT() asm volatile("s_wait_asynccnt 0x0" ::: "memory")

// async 16B global -> LDS transfer (per active lane), tracked by ASYNCcnt
__device__ __forceinline__ void async_ld_b128(const void* lds_ptr, const void* gptr) {
  const unsigned lds = (unsigned)(unsigned long long)(size_t)lds_ptr;  // low 32b of flat = LDS offset
  const unsigned long long ga = (unsigned long long)(size_t)gptr;
  asm volatile("global_load_async_to_lds_b128 %0, %1, off" :: "v"(lds), "v"(ga) : "memory");
}

// ---------------------------------------------------------------------------
// f32 -> f16 elementwise convert (weights, q)
// ---------------------------------------------------------------------------
__global__ __launch_bounds__(TPB)
void cvt_f32_to_f16(const float* __restrict__ src, _Float16* __restrict__ dst, int n) {
  for (int i = blockIdx.x * TPB + threadIdx.x; i < n; i += gridDim.x * TPB)
    dst[i] = (_Float16)src[i];
}

// ---------------------------------------------------------------------------
// Tiled transpose: src [C][S] f32 (per batch) -> dst rows [S][C] f16 with row
// stride dstLd (dst pointer already offset by channel offset for concat).
// ---------------------------------------------------------------------------
__global__ void transpose_to_f16(const float* __restrict__ src, _Float16* __restrict__ dst,
                                 int S, int dstLd, long bsSrc, long bsDst) {
  __shared__ float tile[16][17];
  const int n  = blockIdx.z;
  const int c0 = blockIdx.y << 4, s0 = blockIdx.x << 4;
  const int tx = threadIdx.x, ty = threadIdx.y;
  const float* sp = src + (long)n * bsSrc;
  const int s = s0 + tx;
  tile[ty][tx] = (s < S) ? sp[(long)(c0 + ty) * S + s] : 0.f;
  __syncthreads();
  const int so = s0 + ty;
  if (so < S)
    dst[(long)n * bsDst + (long)so * dstLd + (c0 + tx)] = (_Float16)tile[tx][ty];
}

// ---------------------------------------------------------------------------
// Generic WMMA GEMM: D[m, ncol] = epilogue( sum_k A[m,k] * Bt[ncol,k] )
//   A : f16 row-major M x K (lda = K), M multiple of 256, K multiple of 32
//   Bt: f16 rows = N columns of B, row stride ldb (K-contiguous)
//   out stored transposed: out[ncol * ldout + m]  (f16 if outF16 != null)
//   epilogue: v = acc + bias[m]; if doTanh v = tanh(v); if mul v *= mul[m]
// Block: 256 threads = 8 waves; tile 256(m) x 32(n) x 32(k).
// Wave w owns m rows [32w, 32w+32): 2 A-frags x 2 B-frags -> 4 wmma / k-step.
// Staging uses CDNA5 async global->LDS (ASYNCcnt); LDS double-buffered with
// one barrier per k-step. Epilogue transposes through LDS for b128 stores.
// ---------------------------------------------------------------------------
__global__ __launch_bounds__(TPB)
void gemm_wmma_f16(const _Float16* __restrict__ A, const _Float16* __restrict__ Bt,
                   float* __restrict__ outF32, _Float16* __restrict__ outF16,
                   const float* __restrict__ bias, const float* __restrict__ mul,
                   int M, int N, int K, int ldb, int ldout,
                   long bsB, long bsOut, long bsMul, int doTanh) {
  // rows padded to 40 f16 (80B) to spread LDS banks
  // As: 2 bufs x 256x40 f16 = 40960B ; Bs: 2 bufs x 32x40 f16 = 5120B
  __shared__ __align__(16) char smem[46080];
  _Float16* As = (_Float16*)smem;             // + buf*10240 elements
  _Float16* Bs = (_Float16*)(smem + 40960);   // + buf*1280 elements
  float*    Ef = (float*)smem;                // epilogue tile [32][264] f32
  _Float16* Eh = (_Float16*)smem;             // epilogue tile [32][264] f16

  const int tid  = threadIdx.x;
  const int wave = tid >> 5;
  const int lane = tid & 31;
  const int lo   = lane & 15;
  const int hi   = lane >> 4;
  const int m0   = blockIdx.y << 8;   // * 256
  const int n0   = blockIdx.x << 5;   // * 32
  const int b    = blockIdx.z;

  const _Float16* Ab = A + (long)m0 * K;
  const _Float16* Bb = Bt + (long)b * bsB;

  // stage one 256x32 A tile + 32x32 B tile into buffer `buf` at k-offset k0
  // via async global->LDS b128 transfers (no VGPR round-trip).
  auto stage = [&](int buf, int k0) {
    _Float16* ad = As + buf * 10240;
    #pragma unroll
    for (int i = 0; i < 4; ++i) {
      const int ch  = tid + i * TPB;        // 0..1023
      const int row = ch >> 2;
      const int kc  = (ch & 3) << 3;
      async_ld_b128(ad + row * 40 + kc, Ab + (long)row * K + k0 + kc);
    }
    if (tid < 128) {
      const int row = tid >> 2;
      const int kc  = (tid & 3) << 3;
      // clamp instead of zero-fill: a garbage B column only pollutes the
      // matching output column, which the epilogue never stores.
      int rr = n0 + row; if (rr >= N) rr = N - 1;
      async_ld_b128(Bs + buf * 1280 + row * 40 + kc, Bb + (long)rr * ldb + k0 + kc);
    }
  };

  v8f acc[2][2] = {};

  stage(0, 0);
  ASYNC_WAIT();
  __syncthreads();

  int buf = 0;
  for (int k0 = 0; k0 < K; k0 += 32) {
    if (k0 + 32 < K) stage(buf ^ 1, k0 + 32);

    const _Float16* ab = As + buf * 10240;
    const _Float16* bb = Bs + buf * 1280;
    FragH a[2], bf[2];
    #pragma unroll
    for (int t = 0; t < 2; ++t) {
      const int abase = (32 * wave + 16 * t + lo) * 40 + 8 * hi;
      a[t].f4[0] = *(const float4*)(ab + abase);
      a[t].f4[1] = *(const float4*)(ab + abase + 16);
      const int bbase = (16 * t + lo) * 40 + 8 * hi;
      bf[t].f4[0] = *(const float4*)(bb + bbase);
      bf[t].f4[1] = *(const float4*)(bb + bbase + 16);
    }
    #pragma unroll
    for (int t = 0; t < 2; ++t)
      #pragma unroll
      for (int u = 0; u < 2; ++u)
        acc[t][u] = __builtin_amdgcn_wmma_f32_16x16x32_f16(
            false, a[t].h, false, bf[u].h, (short)0, acc[t][u], false, false);

    ASYNC_WAIT();       // next buffer fully landed in LDS
    __syncthreads();    // all waves done reading current buffer
    buf ^= 1;
  }

  // ---- epilogue: apply bias/tanh/mul, transpose through LDS, store b128 ----
  const float* mulp = mul ? (mul + (long)b * bsMul) : (const float*)0;
  #pragma unroll
  for (int t = 0; t < 2; ++t)
    #pragma unroll
    for (int u = 0; u < 2; ++u)
      #pragma unroll
      for (int r = 0; r < 8; ++r) {
        const int ml = 32 * wave + 16 * t + r + 8 * hi;   // 0..255
        const int nc = 16 * u + lo;                        // 0..31
        float val = acc[t][u][r] + bias[m0 + ml];
        if (doTanh) val = tanhf(val);
        if (mulp)   val *= mulp[m0 + ml];
        if (outF16) Eh[nc * 264 + ml] = (_Float16)val;
        else        Ef[nc * 264 + ml] = val;
      }
  __syncthreads();

  const int  nvalid = (N - n0) < 32 ? (N - n0) : 32;
  const long outB   = (long)b * bsOut;
  if (outF16) {
    for (int i = tid; i < nvalid * 32; i += TPB) {
      const int col = i >> 5;
      const int mc  = (i & 31) << 3;
      *(float4*)(outF16 + outB + (long)(n0 + col) * ldout + m0 + mc) =
          *(const float4*)(Eh + col * 264 + mc);
    }
  } else {
    for (int i = tid; i < nvalid * 64; i += TPB) {
      const int col = i >> 6;
      const int mc  = (i & 63) << 2;
      *(float4*)(outF32 + outB + (long)(n0 + col) * ldout + m0 + mc) =
          *(const float4*)(Ef + col * 264 + mc);
    }
  }
}

// ---------------------------------------------------------------------------
// Per-batch: att = conv2_w @ fused + b2 ; softmax over s ; weighted pooling.
// fused_t: [n][196][512] f32 ; v: [n][2048][196] f32
// outW: [n][2][2048] ; outS: [n][2][196]
// ---------------------------------------------------------------------------
__global__ __launch_bounds__(TPB)
void attn_softmax_pool(const float* __restrict__ fused_t, const float* __restrict__ w2,
                       const float* __restrict__ b2, const float* __restrict__ v,
                       float* __restrict__ outW, float* __restrict__ outS) {
  __shared__ float att[2 * 196];
  __shared__ float soft[2 * 196];
  const int n   = blockIdx.x;
  const int tid = threadIdx.x;

  // att[m][s] = b2[m] + sum_o w2[m][o] * fused_t[n][s][o]
  const float* fb = fused_t + (long)n * 196 * 512;
  for (int idx = tid; idx < 2 * 196; idx += TPB) {
    const int m = idx / 196;
    const int s = idx - m * 196;
    const float* fr = fb + (long)s * 512;
    const float* wr = w2 + m * 512;
    float acc = b2[m];
    for (int o = 0; o < 512; o += 4) {
      const float4 fv = *(const float4*)(fr + o);
      const float4 wv = *(const float4*)(wr + o);
      acc += fv.x * wv.x + fv.y * wv.y + fv.z * wv.z + fv.w * wv.w;
    }
    att[idx] = acc;
  }
  __syncthreads();

  // softmax over s: wave 0 -> m=0, wave 1 -> m=1 (wave32)
  const int wave = tid >> 5, lane = tid & 31;
  if (wave < 2) {
    const int m = wave;
    float mx = -1e30f;
    for (int s = lane; s < 196; s += 32) mx = fmaxf(mx, att[m * 196 + s]);
    for (int o = 16; o > 0; o >>= 1) mx = fmaxf(mx, __shfl_xor(mx, o, 32));
    float sm = 0.f;
    for (int s = lane; s < 196; s += 32) {
      const float e = __expf(att[m * 196 + s] - mx);
      soft[m * 196 + s] = e;
      sm += e;
    }
    for (int o = 16; o > 0; o >>= 1) sm += __shfl_xor(sm, o, 32);
    const float inv = 1.f / sm;
    for (int s = lane; s < 196; s += 32) {
      const float p = soft[m * 196 + s] * inv;
      soft[m * 196 + s] = p;
      outS[(long)n * 392 + m * 196 + s] = p;
    }
  }
  __syncthreads();

  // weighted[m][c] = sum_s soft[m][s] * v[n][c][s]
  const float* vb = v + (long)n * 2048 * 196;
  for (int c = tid; c < 2048; c += TPB) {
    const float* vr = vb + (long)c * 196;
    float a0 = 0.f, a1 = 0.f;
    for (int s = 0; s < 196; ++s) {
      const float f = vr[s];
      a0 += soft[s] * f;
      a1 += soft[196 + s] * f;
    }
    outW[(long)n * 4096 + c]        = a0;
    outW[(long)n * 4096 + 2048 + c] = a1;
  }
}

// ---------------------------------------------------------------------------
// Host launcher
// ---------------------------------------------------------------------------
extern "C" void kernel_launch(void* const* d_in, const int* in_sizes, int n_in,
                              void* d_out, int out_size, void* d_ws, size_t ws_size,
                              hipStream_t stream) {
  const int N = 128, Q = 1024, L = 1024, C = 2048, S = 196;
  const float* q       = (const float*)d_in[0];
  const float* v       = (const float*)d_in[1];
  const float* l       = (const float*)d_in[2];
  const float* lin_w   = (const float*)d_in[3];
  const float* lin_b   = (const float*)d_in[4];
  const float* conv_w  = (const float*)d_in[5];
  const float* conv_b  = (const float*)d_in[6];
  const float* conv1_w = (const float*)d_in[7];
  const float* conv1_b = (const float*)d_in[8];
  const float* conv2_w = (const float*)d_in[9];
  const float* conv2_b = (const float*)d_in[10];

  char* ws = (char*)d_ws;
  size_t off = 0;
  auto take = [&](size_t bytes) { char* p = ws + off; off = (off + bytes + 255) & ~(size_t)255; return p; };
  _Float16* lin_w_h   = (_Float16*)take((size_t)Q * Q * 2);
  _Float16* conv_w_h  = (_Float16*)take((size_t)Q * L * 2);
  _Float16* conv1_w_h = (_Float16*)take((size_t)512 * (Q + C) * 2);
  _Float16* q_h       = (_Float16*)take((size_t)N * Q * 2);
  float*    q1        = (float*)take((size_t)N * Q * 4);
  _Float16* l_t       = (_Float16*)take((size_t)N * S * L * 2);         // [n][s][c]
  _Float16* x_t       = (_Float16*)take((size_t)N * S * (Q + C) * 2);   // [n][s][q | 1024+c]
  float*    fused_t   = (float*)take((size_t)N * S * 512 * 4);          // [n][s][o]
  (void)ws_size; (void)in_sizes; (void)n_in; (void)out_size;

  float* outW = (float*)d_out;                     // [n][2][2048]
  float* outS = (float*)d_out + (size_t)N * 4096;  // [n][2][196]

  auto cvtBlocks = [](int n) { int b = (n + TPB - 1) / TPB; return b > 2048 ? 2048 : b; };

  // 1) convert weights & q to f16
  cvt_f32_to_f16<<<cvtBlocks(Q * Q), TPB, 0, stream>>>(lin_w, lin_w_h, Q * Q);
  cvt_f32_to_f16<<<cvtBlocks(Q * L), TPB, 0, stream>>>(conv_w, conv_w_h, Q * L);
  cvt_f32_to_f16<<<cvtBlocks(512 * (Q + C)), TPB, 0, stream>>>(conv1_w, conv1_w_h, 512 * (Q + C));
  cvt_f32_to_f16<<<cvtBlocks(N * Q), TPB, 0, stream>>>(q, q_h, N * Q);

  // 2) transpose l -> l_t [n][s][c] ; v -> x_t channels [1024,3072)
  {
    dim3 blk(16, 16);
    dim3 grdL((S + 15) / 16, L / 16, N);
    transpose_to_f16<<<grdL, blk, 0, stream>>>(l, l_t, S, L, (long)L * S, (long)S * L);
    dim3 grdV((S + 15) / 16, C / 16, N);
    transpose_to_f16<<<grdV, blk, 0, stream>>>(v, x_t + Q, S, Q + C, (long)C * S, (long)S * (Q + C));
  }

  // 3) q1[b][j] = tanh(q @ lin_w^T + lin_b): A=lin_w_h (M=1024,j), Bt=q_h (N=128 batches)
  {
    dim3 grd(N / 32, Q / 256, 1);
    gemm_wmma_f16<<<grd, TPB, 0, stream>>>(lin_w_h, q_h, q1, (_Float16*)0,
                                           lin_b, (const float*)0,
                                           Q, N, Q, /*ldb*/Q, /*ldout*/Q,
                                           0L, 0L, 0L, /*tanh*/1);
  }

  // 4) f[n][s][q] = q1[n][q] * tanh(conv_w @ l[n] + conv_b) -> x_t channels [0,1024)
  {
    dim3 grd((S + 31) / 32, Q / 256, N);
    gemm_wmma_f16<<<grd, TPB, 0, stream>>>(conv_w_h, l_t, (float*)0, x_t,
                                           conv_b, q1,
                                           Q, S, L, /*ldb*/L, /*ldout*/Q + C,
                                           (long)S * L, (long)S * (Q + C), (long)Q, /*tanh*/1);
  }

  // 5) fused[n][s][o] = conv1_w @ x[n] + conv1_b
  {
    dim3 grd((S + 31) / 32, 512 / 256, N);
    gemm_wmma_f16<<<grd, TPB, 0, stream>>>(conv1_w_h, x_t, fused_t, (_Float16*)0,
                                           conv1_b, (const float*)0,
                                           512, S, Q + C, /*ldb*/Q + C, /*ldout*/512,
                                           (long)S * (Q + C), (long)S * 512, 0L, /*tanh*/0);
  }

  // 6) att -> softmax -> weighted pooling, write both outputs
  attn_softmax_pool<<<N, TPB, 0, stream>>>(fused_t, conv2_w, conv2_b, v, outW, outS);
}
